// LinearW2A16_68676527063117
// MI455X (gfx1250) — matrix operand order repaired and checked by reading, not compiled, not run on hardware
//
#include <hip/hip_runtime.h>

#define M_DIM 4096
#define K_DIM 4096
#define N_DIM 12288
#define GROUP 64

#define BM 128
#define BN 128
#define BK 64
#define NT (K_DIM / BK)   // 64 k-tiles, one quant group each
#define A_STRIDE 72       // padded bf16 row stride -> conflict-free b128 fragment loads
#define B_STRIDE 72

typedef __attribute__((ext_vector_type(16))) __bf16 bf16x16;
typedef __attribute__((ext_vector_type(8)))  __bf16 bf16x8;
typedef __attribute__((ext_vector_type(4)))  __bf16 bf16x4;
typedef __attribute__((ext_vector_type(8)))  float  f32x8;
typedef __attribute__((ext_vector_type(4)))  float  f32x4;
typedef __attribute__((ext_vector_type(4)))  int    i32x4;

__device__ __forceinline__ bf16x16 load_frag(const __bf16* p_lo, const __bf16* p_hi) {
    bf16x8 lo = *(const bf16x8*)p_lo;
    bf16x8 hi = *(const bf16x8*)p_hi;
    return __builtin_shufflevector(lo, hi, 0,1,2,3,4,5,6,7,8,9,10,11,12,13,14,15);
}

__global__ __launch_bounds__(256)
void LinearW2A16_gemm(const float* __restrict__ A,
                      const int*   __restrict__ Q,
                      const float* __restrict__ S,
                      const float* __restrict__ Z,
                      const float* __restrict__ bias,
                      float*       __restrict__ out)
{
    __shared__ __bf16 Alds[2][BM * A_STRIDE];   // [m][k] bf16, double-buffered
    __shared__ __bf16 Blds[2][BN * B_STRIDE];   // [n][k] bf16, K contiguous per column

    const int tid  = threadIdx.x;
    const int lane = tid & 31;
    const int wave = tid >> 5;        // 0..7
    const int wm   = wave >> 2;       // 0..1 : 64 rows of M each
    const int wn   = wave & 3;        // 0..3 : 32 cols of N each
    const int hl   = lane >> 4;       // half-wave select
    const int lcol = lane & 15;

    const int block_m = blockIdx.x * BM;
    const int block_n = blockIdx.y * BN;

    // staging responsibilities (columns fixed per thread)
    const int a_c4 = tid & 15;        // float4 column within BK
    const int a_r0 = tid >> 4;        // A row 0..15, step 16
    const int b_c  = tid & 31;        // int4 column within BN
    const int b_rg = tid >> 5;        // k-row group 0..7 (4 consecutive rows each)
    const int n4   = b_c * 4;

    f32x8 acc[4][2];
#pragma unroll
    for (int i = 0; i < 4; ++i)
#pragma unroll
        for (int j = 0; j < 2; ++j)
            acc[i][j] = (f32x8)0.0f;

    // pipeline registers for the next tile
    f32x4 areg[8];
    i32x4 qreg[8];
    f32x4 sc, zp;

    auto load_tile = [&](int kt) {
        const int k0 = kt * BK;
#pragma unroll
        for (int it = 0; it < 8; ++it)
            areg[it] = *(const f32x4*)(A + (long)(block_m + a_r0 + it * 16) * K_DIM
                                         + k0 + a_c4 * 4);
#pragma unroll
        for (int it = 0; it < 2; ++it)
#pragma unroll
            for (int r = 0; r < 4; ++r)
                qreg[it * 4 + r] = *(const i32x4*)(Q + (long)(k0 + b_rg * 4 + it * 32 + r) * N_DIM
                                                     + block_n + n4);
        sc = *(const f32x4*)(S + (long)kt * N_DIM + block_n + n4);
        zp = *(const f32x4*)(Z + (long)kt * N_DIM + block_n + n4);
    };

    auto store_tile = [&](int buf) {
        // A: fp32 -> bf16, b64 stores
#pragma unroll
        for (int it = 0; it < 8; ++it) {
            bf16x4 h;
            h[0] = (__bf16)areg[it][0]; h[1] = (__bf16)areg[it][1];
            h[2] = (__bf16)areg[it][2]; h[3] = (__bf16)areg[it][3];
            *(bf16x4*)&Alds[buf][(a_r0 + it * 16) * A_STRIDE + a_c4 * 4] = h;
        }
        // B: dequant 4 consecutive K rows per column -> packed b64 stores
#pragma unroll
        for (int it = 0; it < 2; ++it) {
            const int kr = b_rg * 4 + it * 32;
#pragma unroll
            for (int e = 0; e < 4; ++e) {
                bf16x4 h;
#pragma unroll
                for (int r = 0; r < 4; ++r)
                    h[r] = (__bf16)fmaf((float)qreg[it * 4 + r][e], sc[e], zp[e]);
                *(bf16x4*)&Blds[buf][(n4 + e) * B_STRIDE + kr] = h;
            }
        }
    };

    // prologue: stage tile 0
    load_tile(0);
    store_tile(0);

    for (int kt = 0; kt < NT; ++kt) {
        const int buf = kt & 1;

        if (kt + 1 < NT) load_tile(kt + 1);          // HBM loads overlap WMMAs below
        if (kt + 2 < NT) {                           // prefetch tile kt+2 into caches
            const int kp = (kt + 2) * BK;
            __builtin_prefetch(A + (long)(block_m + a_r0) * K_DIM + kp + a_c4 * 4, 0, 1);
            __builtin_prefetch(Q + (long)(kp + b_rg * 4) * N_DIM + block_n + n4, 0, 1);
        }

        __syncthreads();                             // buf is ready for all waves

        const __bf16* __restrict__ Ab = &Alds[buf][0];
        const __bf16* __restrict__ Bb = &Blds[buf][0];
#pragma unroll
        for (int kk = 0; kk < BK; kk += 32) {
            bf16x16 afrag[4];
#pragma unroll
            for (int i = 0; i < 4; ++i) {
                const __bf16* base = &Ab[(wm * 64 + i * 16 + lcol) * A_STRIDE + kk + hl * 8];
                afrag[i] = load_frag(base, base + 16);   // K = kb..kb+7, kb+16..kb+23
            }
            bf16x16 bfrag[2];
#pragma unroll
            for (int j = 0; j < 2; ++j) {
                const __bf16* base = &Bb[(wn * 32 + j * 16 + lcol) * B_STRIDE + kk + hl * 8];
                bfrag[j] = load_frag(base, base + 16);   // K = kb..kb+7, kb+16..kb+23
            }
#pragma unroll
            for (int i = 0; i < 4; ++i)
#pragma unroll
                for (int j = 0; j < 2; ++j)
                    acc[i][j] = __builtin_amdgcn_wmma_f32_16x16x32_bf16(
                        false, afrag[i], false, bfrag[j],
                        (short)0, acc[i][j], false, false);
        }

        if (kt + 1 < NT) store_tile(buf ^ 1);        // fill other buffer; RAW guarded by
                                                     // next iteration's barrier
    }

    // epilogue: C layout (lane -> N, VGPR r -> M) + bias
#pragma unroll
    for (int j = 0; j < 2; ++j) {
        const int n  = block_n + wn * 32 + j * 16 + lcol;
        const float bn = bias[n];
#pragma unroll
        for (int i = 0; i < 4; ++i) {
            const int mbase = block_m + wm * 64 + i * 16 + hl * 8;
#pragma unroll
            for (int r = 0; r < 8; ++r) {
                out[(long)(mbase + r) * N_DIM + n] = acc[i][j][r] + bn;
            }
        }
    }
}

extern "C" void kernel_launch(void* const* d_in, const int* in_sizes, int n_in,
                              void* d_out, int out_size, void* d_ws, size_t ws_size,
                              hipStream_t stream) {
    const float* A    = (const float*)d_in[0];
    const int*   Q    = (const int*)  d_in[1];
    const float* S    = (const float*)d_in[2];
    const float* Z    = (const float*)d_in[3];
    const float* bias = (const float*)d_in[4];
    float* out = (float*)d_out;

    dim3 grid(M_DIM / BM, N_DIM / BN);   // 32 x 96
    dim3 block(256);
    hipLaunchKernelGGL(LinearW2A16_gemm, grid, block, 0, stream,
                       A, Q, S, Z, bias, out);
}
